// SwinTransformerBlock_82265803587928
// MI455X (gfx1250) — compile-verified, matrix operand order checked by reading, not run on hardware
//
#include <hip/hip_runtime.h>
#include <math.h>

// ---------------------------------------------------------------------------
// SwinV2 block for gfx1250 (MI455X): all GEMMs via v_wmma_f32_16x16x32_f16,
// TDM (tensor_load_to_lds) for the proj-kernel activation tile.
// B=32 H=W=56 WS=7 SS=3 C=128 NH=4 HD=32 N=49 nW=64 B_=2048 CPB_HID=512
// ---------------------------------------------------------------------------

typedef __attribute__((ext_vector_type(16))) _Float16 v16h;
typedef __attribute__((ext_vector_type(8)))  _Float16 v8h;
typedef __attribute__((ext_vector_type(8)))  float    v8f;
typedef __attribute__((ext_vector_type(4)))  unsigned int v4u;
typedef __attribute__((ext_vector_type(4)))  int      v4i;
typedef __attribute__((ext_vector_type(8)))  int      v8i;

#define N_WIN   2048          // B * nW
#define NTOK    49
#define CDIM    128
#define HDIM    32

#if __has_builtin(__builtin_amdgcn_tensor_load_to_lds) && \
    __has_builtin(__builtin_amdgcn_s_wait_tensorcnt)
#define HAVE_TDM 1
#else
#define HAVE_TDM 0
#endif

// ---- WMMA helpers ----------------------------------------------------------

__device__ __forceinline__ v8f wmma32(v16h a, v16h b, v8f c) {
  // emits v_wmma_f32_16x16x32_f16
  return __builtin_amdgcn_wmma_f32_16x16x32_f16(false, a, false, b, (short)0, c,
                                                false, false);
}

// K index of fragment element e for this lane (16-bit A/B striping, wave32):
// e0..7 -> K 0..7 (+8 for lanes 16..31), e8..15 -> K 16..23 (+8).
__device__ __forceinline__ int frag_k(int lane, int e) {
  return ((e & 8) ? 16 : 0) + ((lane & 16) ? 8 : 0) + (e & 7);
}

// Load a 16x32 fragment from LDS laid out [row][K], ld in halves (mult of 8).
// Lanes 0..15 hold rows row0..row0+15; K comes in two contiguous 16B chunks,
// so this lowers to two ds_load_b128 per lane.
__device__ __forceinline__ v16h load_frag(const _Float16* base, int row0,
                                          int k0, int ld) {
  const int lane = threadIdx.x & 31;
  const int m  = row0 + (lane & 15);
  const int kb = k0 + ((lane & 16) ? 8 : 0);
  const _Float16* p = base + m * ld + kb;
  v8h lo = *reinterpret_cast<const v8h*>(p);
  v8h hi = *reinterpret_cast<const v8h*>(p + 16);
  v16h a;
#pragma unroll
  for (int e = 0; e < 8; ++e) { a[e] = lo[e]; a[8 + e] = hi[e]; }
  return a;
}

// B fragment straight from a row-major f32 weight matrix [K][ldn] in global
// memory (L2-resident weights), converting to f16 per element.
__device__ __forceinline__ v16h load_frag_gw(const float* __restrict__ w,
                                             int k0, int col, int ldn) {
  const int lane = threadIdx.x & 31;
  const int n = col + (lane & 15);
  v16h b;
#pragma unroll
  for (int e = 0; e < 16; ++e)
    b[e] = (_Float16)w[(size_t)(k0 + frag_k(lane, e)) * ldn + n];
  return b;
}

__device__ __forceinline__ float gelu_exact(float v) {
  return 0.5f * v * (1.0f + erff(v * 0.70710678118654752f));
}

// region id of a *shifted-image* coordinate pair (for the shift mask)
__device__ __forceinline__ int regid(int hg, int wg) {
  int rh = (hg < 49) ? 0 : ((hg < 53) ? 1 : 2);   // H-WS=49, H-SS=53
  int rw = (wg < 49) ? 0 : ((wg < 53) ? 1 : 2);
  return rh * 3 + rw;
}

// ---------------------------------------------------------------------------
// K1: continuous relative position bias MLP -> 16*sigmoid table [169][4]
// ---------------------------------------------------------------------------
__global__ __launch_bounds__(256) void cpb_kernel(
    const float* __restrict__ w1, const float* __restrict__ b1,
    const float* __restrict__ g,  const float* __restrict__ be,
    const float* __restrict__ w2, float* __restrict__ bias16) {
  __shared__ float h1[512];
  __shared__ float red[256];
  const int tid = threadIdx.x;
  const int idx = blockIdx.x;          // 0..168
  const int i = idx / 13, j = idx % 13;

  const float invlog8 = 1.0f / logf(8.0f);
  float ri = (float)(i - 6) * (8.0f / 6.0f);
  float rj = (float)(j - 6) * (8.0f / 6.0f);
  float t0 = copysignf(logf(fabsf(ri) + 1.0f) * invlog8, ri);
  float t1 = copysignf(logf(fabsf(rj) + 1.0f) * invlog8, rj);

  for (int c = tid; c < 512; c += 256)
    h1[c] = t0 * w1[c] + t1 * w1[512 + c] + b1[c];
  __syncthreads();

  // mean
  red[tid] = h1[tid] + h1[tid + 256];
  __syncthreads();
  for (int s = 128; s > 0; s >>= 1) {
    if (tid < s) red[tid] += red[tid + s];
    __syncthreads();
  }
  float mean = red[0] / 512.0f;
  __syncthreads();
  // var
  float d0 = h1[tid] - mean, d1 = h1[tid + 256] - mean;
  red[tid] = d0 * d0 + d1 * d1;
  __syncthreads();
  for (int s = 128; s > 0; s >>= 1) {
    if (tid < s) red[tid] += red[tid + s];
    __syncthreads();
  }
  float inv = rsqrtf(red[0] / 512.0f + 1e-6f);
  __syncthreads();

  for (int c = tid; c < 512; c += 256) {
    float xn = (h1[c] - mean) * inv * g[c] + be[c];
    h1[c] = gelu_exact(xn);
  }
  __syncthreads();

  for (int hh = 0; hh < 4; ++hh) {
    red[tid] = h1[tid] * w2[tid * 4 + hh] + h1[tid + 256] * w2[(tid + 256) * 4 + hh];
    __syncthreads();
    for (int s = 128; s > 0; s >>= 1) {
      if (tid < s) red[tid] += red[tid + s];
      __syncthreads();
    }
    if (tid == 0) bias16[idx * 4 + hh] = 16.0f / (1.0f + expf(-red[0]));
    __syncthreads();
  }
}

// ---------------------------------------------------------------------------
// K2: shift + window partition + QKV GEMM. One workgroup per window (8 waves).
// qkv_out layout (f16): [win][3][head][64][32]
// ---------------------------------------------------------------------------
__global__ __launch_bounds__(256) void qkv_kernel(
    const float* __restrict__ x, const float* __restrict__ qkv_w,
    const float* __restrict__ qkv_b, _Float16* __restrict__ qkv_out) {
  __shared__ alignas(16) _Float16 xs[64 * 136];    // A tile [m][k]
  __shared__ alignas(16) _Float16 wT[128 * 136];   // B tile [n][k] (transposed)
  const int wi = blockIdx.x;
  const int b  = wi >> 6, w_ = wi & 63, wr = w_ >> 3, wc = w_ & 7;
  const int tid = threadIdx.x, wv = tid >> 5, lane = tid & 31;

  // gather shifted window (roll by -SS => source (h+3)%56), zero-pad rows 49..63
  for (int idx = tid; idx < 64 * 128; idx += 256) {
    int m = idx >> 7, c = idx & 127;
    _Float16 v = (_Float16)0.0f;
    if (m < NTOK) {
      int pr = m / 7, pc = m % 7;
      int hg = (wr * 7 + pr + 3) % 56;
      int wg = (wc * 7 + pc + 3) % 56;
      v = (_Float16)x[((size_t)b * 3136 + hg * 56 + wg) * 128 + c];
    }
    xs[m * 136 + c] = v;
  }
  __syncthreads();

  for (int chunk = 0; chunk < 3; ++chunk) {            // 3 * 128 output cols
    for (int idx = tid; idx < 128 * 128; idx += 256) { // wT[n][k] = W[k][n]
      int n = idx >> 7, k = idx & 127;
      wT[n * 136 + k] = (_Float16)qkv_w[(size_t)k * 384 + chunk * 128 + n];
    }
    if (chunk < 2 && tid == 0)
      __builtin_prefetch(qkv_w + (chunk + 1) * 128, 0, 1); // global_prefetch_b8
    __syncthreads();

    const int colT = wv;                                  // 8 col tiles / chunk
    const int nloc = colT * 16 + (lane & 15);
    const int cg   = chunk * 128 + nloc;
    const float bias = qkv_b[cg];
    const int head = (cg & 127) >> 5, d = cg & 31;        // which == chunk
    _Float16* outBase =
        qkv_out + (((size_t)wi * 3 + chunk) * 4 + head) * 2048 + d;

    for (int rt = 0; rt < 4; ++rt) {
      v8f acc = {};
#pragma unroll
      for (int kk = 0; kk < 4; ++kk) {
        v16h a  = load_frag(xs, rt * 16, kk * 32, 136);
        v16h bb = load_frag(wT, colT * 16, kk * 32, 136);
        acc = wmma32(a, bb, acc);
      }
      const int mb = rt * 16 + ((lane & 16) ? 8 : 0);
#pragma unroll
      for (int r = 0; r < 8; ++r) {
        int m = mb + r;
        if (m < NTOK) outBase[m * 32] = (_Float16)(acc[r] + bias);
      }
    }
    __syncthreads();
  }
}

// ---------------------------------------------------------------------------
// K3: cosine attention per (window, head); one wave per workgroup.
// Writes the FULL 64-row O tile (rows >= 49 are true zeros) so the proj
// kernel can bulk-DMA it with the TDM.
// ---------------------------------------------------------------------------
__global__ __launch_bounds__(32) void attn_kernel(
    const _Float16* __restrict__ qkv, const float* __restrict__ logit_scale,
    const float* __restrict__ bias16, _Float16* __restrict__ o_out) {
  __shared__ alignas(16) _Float16 qn[64 * 40];
  __shared__ alignas(16) _Float16 kn[64 * 40];
  __shared__ alignas(16) _Float16 vT[32 * 72];   // [d][m]
  __shared__ alignas(16) float    S[64 * 68];
  __shared__ alignas(16) _Float16 P[64 * 72];

  const int wi = blockIdx.x >> 2, h = blockIdx.x & 3;
  const int w_ = wi & 63, wr = w_ >> 3, wc = w_ & 7;
  const int lane = threadIdx.x;

  const _Float16* qb = qkv + (((size_t)wi * 3 + 0) * 4 + h) * 2048;
  const _Float16* kb = qkv + (((size_t)wi * 3 + 1) * 4 + h) * 2048;
  const _Float16* vb = qkv + (((size_t)wi * 3 + 2) * 4 + h) * 2048;

  for (int r = lane; r < 64; r += 32) {
    if (r < NTOK) {
      float t[32], ss;
      ss = 0.0f;
#pragma unroll
      for (int d = 0; d < 32; ++d) { t[d] = (float)qb[r * 32 + d]; ss += t[d] * t[d]; }
      float rn = rsqrtf(ss + 1e-12f);
#pragma unroll
      for (int d = 0; d < 32; ++d) qn[r * 40 + d] = (_Float16)(t[d] * rn);
      ss = 0.0f;
#pragma unroll
      for (int d = 0; d < 32; ++d) { t[d] = (float)kb[r * 32 + d]; ss += t[d] * t[d]; }
      rn = rsqrtf(ss + 1e-12f);
#pragma unroll
      for (int d = 0; d < 32; ++d) kn[r * 40 + d] = (_Float16)(t[d] * rn);
#pragma unroll
      for (int d = 0; d < 32; ++d) vT[d * 72 + r] = vb[r * 32 + d];
    } else {
#pragma unroll
      for (int d = 0; d < 32; ++d) {
        qn[r * 40 + d] = (_Float16)0.0f;
        kn[r * 40 + d] = (_Float16)0.0f;
        vT[d * 72 + r] = (_Float16)0.0f;
      }
    }
  }
  __syncthreads();

  const float scale =
      expf(fminf(fmaxf(logit_scale[h], -100.0f), logf(100.0f)));

  // S = qn @ kn^T  (+ scale + CPB bias + shift mask)
  for (int rt = 0; rt < 4; ++rt) {
    for (int ct = 0; ct < 4; ++ct) {
      v8f acc = {};
      v16h a  = load_frag(qn, rt * 16, 0, 40);
      v16h bb = load_frag(kn, ct * 16, 0, 40);   // kn [m][d] is exactly B [n][k]
      acc = wmma32(a, bb, acc);

      const int j  = ct * 16 + (lane & 15);
      const int mb = rt * 16 + ((lane & 16) ? 8 : 0);
      const bool jok = j < NTOK;
      int jh = 0, jw = 0, cj = 0;
      if (jok) { jh = j / 7; jw = j % 7; cj = regid(wr * 7 + jh, wc * 7 + jw); }
#pragma unroll
      for (int r = 0; r < 8; ++r) {
        int i = mb + r;
        float val = -1e30f;
        if (jok && i < NTOK) {
          int ih = i / 7, iw = i % 7;
          int ridx = (ih - jh + 6) * 13 + (iw - jw + 6);
          float bias = bias16[ridx * 4 + h];
          float msk  = (regid(wr * 7 + ih, wc * 7 + iw) != cj) ? -100.0f : 0.0f;
          val = acc[r] * scale + bias + msk;
        }
        S[i * 68 + j] = val;
      }
    }
  }
  __syncthreads();

  // row softmax (f32), write P as f16 padded to 64 cols
  for (int r = lane; r < 64; r += 32) {
    if (r < NTOK) {
      float mx = -1e30f;
      for (int j = 0; j < NTOK; ++j) mx = fmaxf(mx, S[r * 68 + j]);
      float sum = 0.0f;
      for (int j = 0; j < NTOK; ++j) {
        float e = __expf(S[r * 68 + j] - mx);
        S[r * 68 + j] = e;
        sum += e;
      }
      float inv = 1.0f / sum;
      for (int j = 0; j < NTOK; ++j) P[r * 72 + j] = (_Float16)(S[r * 68 + j] * inv);
      for (int j = NTOK; j < 64; ++j) P[r * 72 + j] = (_Float16)0.0f;
    } else {
      for (int j = 0; j < 64; ++j) P[r * 72 + j] = (_Float16)0.0f;
    }
  }
  __syncthreads();

  // O = P @ v   -> o_out [win][m][h*32+d] (f16); all 64 rows written
  for (int rt = 0; rt < 4; ++rt) {
    for (int ct = 0; ct < 2; ++ct) {
      v8f acc = {};
#pragma unroll
      for (int kk = 0; kk < 2; ++kk) {
        v16h a  = load_frag(P, rt * 16, kk * 32, 72);
        v16h bb = load_frag(vT, ct * 16, kk * 32, 72);
        acc = wmma32(a, bb, acc);
      }
      const int dcol = ct * 16 + (lane & 15);
      const int mb   = rt * 16 + ((lane & 16) ? 8 : 0);
#pragma unroll
      for (int r = 0; r < 8; ++r) {
        int m = mb + r;
        o_out[((size_t)wi * 64 + m) * 128 + h * 32 + dcol] = (_Float16)acc[r];
      }
    }
  }
}

// ---------------------------------------------------------------------------
// K4: output projection + window reverse + un-shift scatter (f32).
// The 16KB contiguous activation tile is DMA'd into LDS by the Tensor Data
// Mover (with pad_interval/pad_amount generating the 136-half row stride)
// while all 256 threads transpose-convert the proj weights.
// ---------------------------------------------------------------------------
__global__ __launch_bounds__(256) void proj_kernel(
    const _Float16* __restrict__ o_in, const float* __restrict__ proj_w,
    const float* __restrict__ proj_b, float* __restrict__ proj_out) {
  __shared__ alignas(16) _Float16 os[64 * 136];
  __shared__ alignas(16) _Float16 wT[128 * 136];
  const int wi = blockIdx.x;
  const int b  = wi >> 6, w_ = wi & 63, wr = w_ >> 3, wc = w_ & 7;
  const int tid = threadIdx.x, wv = tid >> 5, lane = tid & 31;

#if HAVE_TDM
  if (wv == 0) {
    // ---- Tensor DMA descriptor: 1-D tile of 8192 f16 with LDS padding of
    // 4 DWORDs after every 64 DWORDs (row stride 136 halves). ----
    unsigned long long ga =
        (unsigned long long)(const void*)(o_in + (size_t)wi * 8192);
    unsigned lds = (unsigned)(unsigned long long)(const void*)os;
    v4u g0;
    g0[0] = 1u;                                        // count=1, user mode
    g0[1] = lds;                                       // lds_addr
    g0[2] = (unsigned)(ga & 0xFFFFFFFFu);              // global_addr[31:0]
    g0[3] = (unsigned)((ga >> 32) & 0x01FFFFFFu) | 0x80000000u; // [56:32]|type=2
    v8i g1;
    g1[0] = (int)((1u << 16)        // data_size = 2 bytes
                | (1u << 20)        // pad_enable
                | (5u << 22)        // pad_interval: 64 DWORDs
                | (3u << 25));      // pad_amount:   4 DWORDs
    g1[1] = (int)(8192u << 16);     // tensor_dim0[15:0] (=8192) in [63:48]
    g1[2] = (int)(1u << 16);        // tensor_dim0[31:16]=0 | tensor_dim1=1
    g1[3] = (int)(8192u << 16);     // tensor_dim1 hi=0 | tile_dim0 = 8192
    g1[4] = 1;                      // tile_dim1 = 1, tile_dim2 = 0
    g1[5] = 8192;                   // tensor_dim0_stride[31:0]
    g1[6] = 0;
    g1[7] = 0;
    v4i z4 = {0, 0, 0, 0};
#if defined(__clang_major__) && (__clang_major__ >= 23)
    v8i z8 = {0, 0, 0, 0, 0, 0, 0, 0};
    __builtin_amdgcn_tensor_load_to_lds(g0, g1, z4, z4, z8, 0);
#else
    __builtin_amdgcn_tensor_load_to_lds(g0, g1, z4, z4, 0);
#endif
  }
#else
  for (int idx = tid; idx < 64 * 128; idx += 256)
    os[(idx >> 7) * 136 + (idx & 127)] = o_in[(size_t)wi * 8192 + idx];
#endif

  // transpose-convert weights while the TDM streams the activation tile
  for (int idx = tid; idx < 128 * 128; idx += 256) {
    int n = idx >> 7, k = idx & 127;
    wT[n * 136 + k] = (_Float16)proj_w[(size_t)k * 128 + n];
  }
#if HAVE_TDM
  if (wv == 0) __builtin_amdgcn_s_wait_tensorcnt(0);
#endif
  __syncthreads();

  const int ct  = wv;
  const int col = ct * 16 + (lane & 15);
  const float bias = proj_b[col];
  for (int rt = 0; rt < 4; ++rt) {
    v8f acc = {};
#pragma unroll
    for (int kk = 0; kk < 4; ++kk) {
      v16h a  = load_frag(os, rt * 16, kk * 32, 136);
      v16h bb = load_frag(wT, ct * 16, kk * 32, 136);
      acc = wmma32(a, bb, acc);
    }
    const int mb = rt * 16 + ((lane & 16) ? 8 : 0);
#pragma unroll
    for (int r = 0; r < 8; ++r) {
      int m = mb + r;
      if (m < NTOK) {
        int pr = m / 7, pc = m % 7;
        int hg = (wr * 7 + pr + 3) % 56;   // un-shift: roll(+SS)
        int wg = (wc * 7 + pc + 3) % 56;
        proj_out[((size_t)b * 3136 + hg * 56 + wg) * 128 + col] = acc[r] + bias;
      }
    }
  }
}

// ---------------------------------------------------------------------------
// K5: x1 = x + LN1(proj); m = gelu(x1@W1+b1)@W2+b2; out = x1 + LN2(m)
// 64 tokens per workgroup; hidden (512) processed in 4 chunks of 128 with
// WMMA output accumulators held in VGPRs across chunks. Weights fetched as
// fragments directly from (L2-resident) global memory.
// ---------------------------------------------------------------------------
__global__ __launch_bounds__(256) void mlp_kernel(
    const float* __restrict__ x, const float* __restrict__ proj,
    const float* __restrict__ n1g, const float* __restrict__ n1b,
    const float* __restrict__ w1, const float* __restrict__ b1,
    const float* __restrict__ w2, const float* __restrict__ b2,
    const float* __restrict__ n2g, const float* __restrict__ n2b,
    float* __restrict__ out) {
  __shared__ alignas(16) _Float16 x1h[64 * 136];
  __shared__ alignas(16) _Float16 hh[64 * 136];
  __shared__ alignas(16) _Float16 mm[64 * 136];
  const int tid = threadIdx.x, wv = tid >> 5, lane = tid & 31;
  const size_t tb = (size_t)blockIdx.x * 64;

  if (tid < 64) {  // row owner: LN1 + residual -> x1 (f16)
    const float* pr = proj + (tb + tid) * 128;
    const float* xr = x + (tb + tid) * 128;
    float mean = 0.0f;
    for (int c = 0; c < 128; ++c) mean += pr[c];
    mean *= (1.0f / 128.0f);
    float var = 0.0f;
    for (int c = 0; c < 128; ++c) { float d = pr[c] - mean; var += d * d; }
    float inv = rsqrtf(var * (1.0f / 128.0f) + 1e-6f);
    for (int c = 0; c < 128; ++c)
      x1h[tid * 136 + c] =
          (_Float16)(xr[c] + (pr[c] - mean) * inv * n1g[c] + n1b[c]);
  }
  __syncthreads();

  const int rt = wv >> 1;          // row tile of this wave
  const int cb = (wv & 1) * 4;     // first of 4 col tiles of this wave
  v8f acc[4] = {};

  for (int chunk = 0; chunk < 4; ++chunk) {
    // hidden chunk: h = gelu(x1 @ W1[:, chunk]) -> hh (f16)
#pragma unroll
    for (int ctl = 0; ctl < 4; ++ctl) {
      const int ct  = cb + ctl;
      const int ngl = chunk * 128 + ct * 16;
      v8f ha = {};
#pragma unroll
      for (int kk = 0; kk < 4; ++kk) {
        v16h a  = load_frag(x1h, rt * 16, kk * 32, 136);
        v16h bb = load_frag_gw(w1, kk * 32, ngl, 512);
        ha = wmma32(a, bb, ha);
      }
      const float bias = b1[ngl + (lane & 15)];
      const int mb = rt * 16 + ((lane & 16) ? 8 : 0);
      const int cl = ct * 16 + (lane & 15);
#pragma unroll
      for (int r = 0; r < 8; ++r)
        hh[(mb + r) * 136 + cl] = (_Float16)gelu_exact(ha[r] + bias);
    }
    __syncthreads();

    // accumulate out tiles over this hidden chunk
#pragma unroll
    for (int ctl = 0; ctl < 4; ++ctl) {
      const int oc = cb + ctl;
#pragma unroll
      for (int kk = 0; kk < 4; ++kk) {
        v16h a  = load_frag(hh, rt * 16, kk * 32, 136);
        v16h bb = load_frag_gw(w2, chunk * 128 + kk * 32, oc * 16, 128);
        acc[ctl] = wmma32(a, bb, acc[ctl]);
      }
    }
    __syncthreads();
  }

  // stage m (+b2) into LDS for LN2
#pragma unroll
  for (int ctl = 0; ctl < 4; ++ctl) {
    const int oc  = cb + ctl;
    const int col = oc * 16 + (lane & 15);
    const float bias = b2[col];
    const int mb = rt * 16 + ((lane & 16) ? 8 : 0);
#pragma unroll
    for (int r = 0; r < 8; ++r)
      mm[(mb + r) * 136 + col] = (_Float16)(acc[ctl][r] + bias);
  }
  __syncthreads();

  if (tid < 64) {  // row owner: out = x1 + LN2(m)
    float mean = 0.0f;
    for (int c = 0; c < 128; ++c) mean += (float)mm[tid * 136 + c];
    mean *= (1.0f / 128.0f);
    float var = 0.0f;
    for (int c = 0; c < 128; ++c) {
      float d = (float)mm[tid * 136 + c] - mean;
      var += d * d;
    }
    float inv = rsqrtf(var * (1.0f / 128.0f) + 1e-6f);
    float* o = out + (tb + tid) * 128;
    for (int c = 0; c < 128; ++c)
      o[c] = (float)x1h[tid * 136 + c] +
             ((float)mm[tid * 136 + c] - mean) * inv * n2g[c] + n2b[c];
  }
}

// ---------------------------------------------------------------------------
// launch
// ---------------------------------------------------------------------------
extern "C" void kernel_launch(void* const* d_in, const int* in_sizes, int n_in,
                              void* d_out, int out_size, void* d_ws,
                              size_t ws_size, hipStream_t stream) {
  (void)in_sizes; (void)n_in; (void)out_size; (void)ws_size;
  const float* x        = (const float*)d_in[0];
  const float* qkv_w    = (const float*)d_in[1];
  const float* qkv_b    = (const float*)d_in[2];
  const float* proj_w   = (const float*)d_in[3];
  const float* proj_b   = (const float*)d_in[4];
  const float* lscale   = (const float*)d_in[5];
  const float* cpb_w1   = (const float*)d_in[6];
  const float* cpb_b1   = (const float*)d_in[7];
  const float* cpb_g    = (const float*)d_in[8];
  const float* cpb_be   = (const float*)d_in[9];
  const float* cpb_w2   = (const float*)d_in[10];
  const float* norm1_g  = (const float*)d_in[11];
  const float* norm1_b  = (const float*)d_in[12];
  const float* norm2_g  = (const float*)d_in[13];
  const float* norm2_b  = (const float*)d_in[14];
  const float* mlp_w1   = (const float*)d_in[15];
  const float* mlp_b1   = (const float*)d_in[16];
  const float* mlp_w2   = (const float*)d_in[17];
  const float* mlp_b2   = (const float*)d_in[18];
  float* out = (float*)d_out;

  // workspace layout (bytes):
  //   [0,4096)                    bias16 table (169*4 f32)
  //   [4096, +100663296)          qkv f16  [2048][3][4][64][32]
  //   next  +33554432             attn-out f16 [2048][64][128]
  //   next  +51380224             proj-out f32 [32][3136][128]
  char* ws = (char*)d_ws;
  float*    bias16 = (float*)ws;
  _Float16* qkv    = (_Float16*)(ws + 4096);
  _Float16* o_buf  = (_Float16*)(ws + 4096 + 100663296u);
  float*    projo  = (float*)(ws + 4096 + 100663296u + 33554432u);

  cpb_kernel<<<169, 256, 0, stream>>>(cpb_w1, cpb_b1, cpb_g, cpb_be, cpb_w2,
                                      bias16);
  qkv_kernel<<<N_WIN, 256, 0, stream>>>(x, qkv_w, qkv_b, qkv);
  attn_kernel<<<N_WIN * 4, 32, 0, stream>>>(qkv, lscale, bias16, o_buf);
  proj_kernel<<<N_WIN, 256, 0, stream>>>(o_buf, proj_w, proj_b, projo);
  mlp_kernel<<<(32 * 3136) / 64, 256, 0, stream>>>(
      x, projo, norm1_g, norm1_b, mlp_w1, mlp_b1, mlp_w2, mlp_b2, norm2_g,
      norm2_b, out);
}
